// DynamicConv_64226940944389
// MI455X (gfx1250) — matrix operand order repaired and checked by reading, not compiled, not run on hardware
//
#include <hip/hip_runtime.h>
#include <hip/hip_bf16.h>

typedef __attribute__((ext_vector_type(16))) __bf16 v16bf;
typedef __attribute__((ext_vector_type(8)))  __bf16 v8bf;
typedef __attribute__((ext_vector_type(8)))  float  v8f;
typedef __attribute__((ext_vector_type(4)))  unsigned int v4u;
typedef __attribute__((ext_vector_type(8)))  int v8i;
typedef __attribute__((ext_vector_type(4)))  int v4i;

#define B_    32
#define C_    256
#define HW_   56
#define NPIX  (56 * 56)
#define K_    4
#define O_    256
#define HID   65
#define KTOT  2304            // C * 9
#define BANK  589824          // O * C * 9 (one weight bank)
#define WROWS 576             // 64 o * 9 taps, rows per weight tile
#define WSTRIDE 40            // LDS row stride in halfwords (80 B)

// ---------------- kernel 1: global average pool -> pooled[B][C] ----------------
__global__ void pool_kernel(const float* __restrict__ x, float* __restrict__ pooled) {
    int gwid = (blockIdx.x * blockDim.x + threadIdx.x) >> 5;   // one wave per (b,c)
    int lane = threadIdx.x & 31;
    if (gwid >= B_ * C_) return;
    const float* p = x + (size_t)gwid * NPIX;
    float s = 0.f;
    for (int i = lane; i < NPIX; i += 32) s += p[i];
    #pragma unroll
    for (int off = 16; off > 0; off >>= 1) s += __shfl_xor(s, off, 32);
    if (lane == 0) pooled[gwid] = s * (1.0f / (float)NPIX);
}

// ---------------- kernel 2: fc1 -> relu -> fc2 -> softmax -> atten[B][K] --------
__global__ void atten_kernel(const float* __restrict__ pooled,
                             const float* __restrict__ fc1,
                             const float* __restrict__ fc2,
                             float* __restrict__ atten) {
    int b = threadIdx.x;
    if (b >= B_) return;
    const float* pb = pooled + b * C_;
    float logit[K_] = {0.f, 0.f, 0.f, 0.f};
    for (int j = 0; j < HID; ++j) {
        const float* f1 = fc1 + j * C_;
        float s = 0.f;
        for (int c = 0; c < C_; ++c) s += pb[c] * f1[c];
        s = fmaxf(s, 0.f);
        #pragma unroll
        for (int k = 0; k < K_; ++k) logit[k] += s * fc2[k * HID + j];
    }
    float m = logit[0];
    #pragma unroll
    for (int k = 1; k < K_; ++k) m = fmaxf(m, logit[k]);
    float e[K_], se = 0.f;
    #pragma unroll
    for (int k = 0; k < K_; ++k) { e[k] = __expf(logit[k] - m); se += e[k]; }
    float inv = 1.0f / se;
    #pragma unroll
    for (int k = 0; k < K_; ++k) atten[b * K_ + k] = e[k] * inv;
}

// ---------------- kernel 3: mix banks, transpose (c,r)->(r,c), cast bf16 --------
// aggw layout: [b][o][tap r][c]  (c contiguous -> TDM rows / WMMA A-fragments)
__global__ __launch_bounds__(256) void mix_kernel(const float* __restrict__ weight,
                                                  const float* __restrict__ atten,
                                                  __hip_bfloat16* __restrict__ aggw) {
    __shared__ __align__(8) __hip_bfloat16 lw[KTOT];
    int b = blockIdx.x >> 8;        // grid = B*O blocks
    int o = blockIdx.x & 255;
    float a0 = atten[b * K_ + 0], a1 = atten[b * K_ + 1];
    float a2 = atten[b * K_ + 2], a3 = atten[b * K_ + 3];
    const float* w0 = weight + (size_t)o * KTOT;     // reference order: [o][c][r]
    for (int t = threadIdx.x; t < KTOT; t += 256) {  // coalesced reads over t=c*9+r
        float s = a0 * w0[t] + a1 * w0[t + BANK] + a2 * w0[t + 2 * BANK] + a3 * w0[t + 3 * BANK];
        int c = t / 9, r = t - c * 9;
        lw[r * C_ + c] = __float2bfloat16(s);        // transpose through LDS
    }
    __syncthreads();
    uint2* dst = (uint2*)(aggw + (size_t)blockIdx.x * KTOT);
    const uint2* src = (const uint2*)lw;
    for (int t = threadIdx.x; t < KTOT / 4; t += 256) dst[t] = src[t];  // coalesced writes
}

// ---------------- TDM: stage one 576-row x 32-bf16 weight slice into LDS --------
// LDS rows get 16 B of padding after every 64 B row -> 80 B row stride.
__device__ __forceinline__ void tdm_load_weights(const __hip_bfloat16* gptr,
                                                 unsigned lds_off) {
    unsigned long long ga = (unsigned long long)(uintptr_t)gptr;
    v4u g0 = { 1u,                                   // count=1, user mode
               lds_off,                               // lds_addr [63:32]
               (unsigned)(ga & 0xFFFFFFFFu),          // global_addr[31:0]
               (unsigned)((ga >> 32) & 0x01FFFFFFu) | (2u << 30) }; // addr[56:32] | type=2
    v8i g1 = { (int)((1u << 16) | (1u << 20) | (3u << 22) | (3u << 25)),
               // ^ data_size=1 (2B), pad_enable, pad_interval=3 (16 DW), pad_amount=3 (4 DW)
               (int)(32u << 16),          // tensor_dim0[15:0]=32 in bits[63:48]
               (int)(576u << 16),         // tensor_dim0 hi=0; tensor_dim1[15:0]=576
               (int)(32u << 16),          // tensor_dim1 hi=0; tile_dim0=32
               (int)(576u),               // tile_dim1=576; tile_dim2=0
               (int)(256u),               // tensor_dim0_stride[31:0]=256 elements
               0, 0 };                    // stride hi / dim1_stride = 0
    v4i gz = { 0, 0, 0, 0 };
    v8i gz8 = { 0, 0, 0, 0, 0, 0, 0, 0 };
    __builtin_amdgcn_tensor_load_to_lds(g0, g1, gz, gz, gz8, 0);
}

// ---------------- kernel 4: per-sample conv as 9-tap implicit GEMM (WMMA) -------
// block tile: 64 output channels x 64 pixels (exact 8x8 patch), 256 threads = 8 waves
// weights DMA'd by TDM (double buffered), x halo staged + converted by VALU.
__global__ __launch_bounds__(256, 1) void conv_kernel(const float* __restrict__ x,
                                                      const __hip_bfloat16* __restrict__ aggw,
                                                      float* __restrict__ out) {
    const int b   = blockIdx.z;
    const int o0  = blockIdx.y * 64;
    const int ph0 = (blockIdx.x / 7) * 8;
    const int pw0 = (blockIdx.x % 7) * 8;

    // xs: [pos 0..99][c 0..31], row stride 40 halfwords (80B)
    __shared__ __align__(16) __hip_bfloat16 xs[100 * WSTRIDE];
    // ws2: two TDM buffers, rows in global order (o*9 + r), stride 40 halfwords
    __shared__ __align__(16) __hip_bfloat16 ws2[2][WROWS * WSTRIDE];

    const int tid  = threadIdx.x;
    const int lane = tid & 31;
    const int wave = tid >> 5;
    const int l16  = lane & 15;
    const int half = lane >> 4;
    const int mt   = wave & 3;
    const int ntp  = wave >> 2;

    v8f acc0 = {}; v8f acc1 = {};

    const float* xb = x + (size_t)b * C_ * NPIX;
    const __hip_bfloat16* wgb = aggw + (size_t)(b * O_ + o0) * KTOT;

    const int oi  = mt * 16 + l16;      // A-fragment row (output channel in tile)
    const int pl0 = ntp * 32 + l16;     // B-fragment cols (pixels in tile)
    const int pl1 = pl0 + 16;

    // prologue: DMA weights for c-block 0 into buffer 0
    if (wave == 0) {
        tdm_load_weights(wgb, (unsigned)(uintptr_t)&ws2[0][0]);
    }

    for (int cb = 0; cb < 8; ++cb) {    // 8 blocks of 32 channels
        __syncthreads();                // prev iter done reading xs / ws buffers
        // ---- stage 10x10 halo patch of x for 32 channels, f32 -> bf16 ----
        for (int task = tid; task < 320; task += 256) {
            int c = task / 10;
            int i = task - c * 10;
            int h = ph0 - 1 + i;
            const float* xrow = xb + ((size_t)(cb * 32 + c) * HW_ + h) * HW_;
            #pragma unroll
            for (int j = 0; j < 10; ++j) {
                int w = pw0 - 1 + j;
                float v = (h >= 0 && h < HW_ && w >= 0 && w < HW_) ? xrow[w] : 0.0f;
                xs[(i * 10 + j) * WSTRIDE + c] = __float2bfloat16(v);
            }
        }
        // ---- prefetch next weight slice, then wait for current one ----
        if (wave == 0) {
            if (cb < 7) {
                tdm_load_weights(wgb + (cb + 1) * 32,
                                 (unsigned)(uintptr_t)&ws2[(cb + 1) & 1][0]);
                __builtin_amdgcn_s_wait_tensorcnt(1);   // current buffer ready
            } else {
                __builtin_amdgcn_s_wait_tensorcnt(0);
            }
        }
        __syncthreads();                // xs staged + ws[cb&1] DMA complete

        const __hip_bfloat16* wsb = &ws2[cb & 1][0];
        #pragma unroll
        for (int r = 0; r < 9; ++r) {
            const int dh = r / 3, dw = r % 3;
            // A fragment (weights 16x32): LDS row = oi*9 + r (TDM row order)
            const __hip_bfloat16* wrow = wsb + (oi * 9 + r) * WSTRIDE;
            v8bf alo = *(const v8bf*)(wrow + 8 * half);
            v8bf ahi = *(const v8bf*)(wrow + 16 + 8 * half);
            v16bf a = __builtin_shufflevector(alo, ahi,
                0,1,2,3,4,5,6,7,8,9,10,11,12,13,14,15);
            // B fragments (shifted pixels 32x16): c = 8*half + s, then 16 + 8*half + s
            int pos0 = ((pl0 >> 3) + dh) * 10 + (pl0 & 7) + dw;
            int pos1 = ((pl1 >> 3) + dh) * 10 + (pl1 & 7) + dw;
            v8bf b0l = *(const v8bf*)(&xs[pos0 * WSTRIDE + 8 * half]);
            v8bf b0h = *(const v8bf*)(&xs[pos0 * WSTRIDE + 16 + 8 * half]);
            v8bf b1l = *(const v8bf*)(&xs[pos1 * WSTRIDE + 8 * half]);
            v8bf b1h = *(const v8bf*)(&xs[pos1 * WSTRIDE + 16 + 8 * half]);
            v16bf bf0 = __builtin_shufflevector(b0l, b0h,
                0,1,2,3,4,5,6,7,8,9,10,11,12,13,14,15);
            v16bf bf1 = __builtin_shufflevector(b1l, b1h,
                0,1,2,3,4,5,6,7,8,9,10,11,12,13,14,15);

            acc0 = __builtin_amdgcn_wmma_f32_16x16x32_bf16(
                false, a, false, bf0, (short)0, acc0, false, false);
            acc1 = __builtin_amdgcn_wmma_f32_16x16x32_bf16(
                false, a, false, bf1, (short)0, acc1, false, false);
        }
    }

    // ---- store: D vgpr rr -> M = mt*16 + 8*half + rr (o), N = l16 (pixel) ----
    const int h0a = ph0 + (pl0 >> 3), w0a = pw0 + (pl0 & 7);
    const int h1a = ph0 + (pl1 >> 3), w1a = pw0 + (pl1 & 7);
    #pragma unroll
    for (int rr = 0; rr < 8; ++rr) {
        int o = o0 + mt * 16 + 8 * half + rr;
        float* ob = out + (size_t)(b * O_ + o) * NPIX;
        ob[h0a * HW_ + w0a] = acc0[rr];
        ob[h1a * HW_ + w1a] = acc1[rr];
    }
}

// ---------------- launcher ----------------
extern "C" void kernel_launch(void* const* d_in, const int* in_sizes, int n_in,
                              void* d_out, int out_size, void* d_ws, size_t ws_size,
                              hipStream_t stream) {
    const float* x      = (const float*)d_in[0];
    const float* fc1    = (const float*)d_in[1];
    const float* fc2    = (const float*)d_in[2];
    const float* weight = (const float*)d_in[3];
    float* out = (float*)d_out;

    // workspace: pooled (32KB) | atten (512B) | aggw bf16 (36MB)
    float* pooled = (float*)d_ws;
    float* atten  = pooled + B_ * C_;
    __hip_bfloat16* aggw = (__hip_bfloat16*)(atten + B_ * K_);

    pool_kernel<<<(B_ * C_) / 8, 256, 0, stream>>>(x, pooled);
    atten_kernel<<<1, 32, 0, stream>>>(pooled, fc1, fc2, atten);
    mix_kernel<<<B_ * O_, 256, 0, stream>>>(weight, atten, aggw);
    conv_kernel<<<dim3(49, 4, B_), 256, 0, stream>>>(x, aggw, out);
}
